// TGAE_Encoder_19963007992406
// MI455X (gfx1250) — matrix operand order repaired and checked by reading, not compiled, not run on hardware
//
#include <hip/hip_runtime.h>

#define NNODES   50000
#define IN_DIM   256
#define HID      512
#define NLAYERS  3
#define OUT_DIM  256
#define LDSW     20     // LDS row stride in dwords: 16B-aligned rows, conflict-free

typedef __attribute__((ext_vector_type(16))) __bf16 v16bf;
typedef __attribute__((ext_vector_type(8)))  float  v8f;

union Frag16 {
    v16bf bf;
    unsigned int u32[8];
    uint4 q[2];
};

// Pack two fp32 -> bf16x2 dword: round (+0x8000) then v_perm_b32 byte select.
// result[15:0] = bf16(lo), result[31:16] = bf16(hi)   (3 VALU ops total)
__device__ __forceinline__ unsigned int pk_bf16(float lo, float hi) {
    union { float f; unsigned int u; } a, b;
    a.f = lo; b.f = hi;
    return __builtin_amdgcn_perm(b.u + 0x8000u, a.u + 0x8000u, 0x07060302u);
}

// ---------------------------------------------------------------------------
// Tiled mixed-precision GEMM: C[M,N] = op(A [+ A2]) @ W + bias, optional ReLU,
// optional beta=1 accumulation. fp32 in memory, bf16 in LDS, fp32 accum via
// v_wmma_f32_16x16x32_bf16. 256 threads = 8 wave32s; 128x128 block tile;
// wave tile 32x64 = 2x4 WMMA tiles. Double-buffered LDS, one barrier/K-step,
// fragments loaded with ds_load_b128 (20-dword rows keep 16B alignment and
// tile all 64 banks without conflict).
// ---------------------------------------------------------------------------
template <bool ADD2, bool RELU, bool ACCUM>
__global__ __launch_bounds__(256)
void gemm_bf16_wmma(const float* __restrict__ A, const float* __restrict__ A2,
                    const float* __restrict__ W, const float* __restrict__ bias,
                    float* __restrict__ C, int M, int N, int K)
{
    __shared__ __align__(16) unsigned int sA[2][128 * LDSW];  // [buf][row][k/2] bf16x2
    __shared__ __align__(16) unsigned int sB[2][128 * LDSW];  // [buf][col][k/2] bf16x2 (transposed)

    const int tid   = threadIdx.x;
    const int n_blk = blockIdx.x * 128;
    const int m_blk = blockIdx.y * 128;

    const int wave = tid >> 5;   // wave32
    const int lane = tid & 31;
    const int wm   = wave >> 1;  // 0..3: 32-row slab
    const int wn   = wave & 1;   // 0..1: 64-col slab
    const int lr   = lane & 15;  // row/col within 16x16 tile
    const int kh   = lane >> 4;  // K-half select

    v8f acc[2][4] = {};
    float4 ra[4], ra2[4], rb[2][2];

    // ---- stage global -> registers for K-tile at k0 ----
    auto load_tiles = [&](int k0) {
#pragma unroll
        for (int r = 0; r < 4; ++r) {             // A: 128 rows x 32 k
            int id   = tid + 256 * r;
            int row  = id >> 3;
            int k4   = (id & 7) << 2;
            int grow = m_blk + row;
            ra[r] = make_float4(0.f, 0.f, 0.f, 0.f);
            if (ADD2) ra2[r] = make_float4(0.f, 0.f, 0.f, 0.f);
            if (grow < M) {
                const float* ap = A + (size_t)grow * K + k0 + k4;
                ra[r] = *(const float4*)ap;
                if (ADD2) ra2[r] = *(const float4*)(A2 + (size_t)grow * K + k0 + k4);
                if (r == 0 && k0 + 32 < K)
                    __builtin_prefetch(ap + 32, 0, 3);   // WGP-scope global_prefetch_b8
            }
        }
#pragma unroll
        for (int r = 0; r < 2; ++r) {             // B: 32 k x 128 n (k-pairs)
            int id  = tid + 256 * r;              // 0..511 pair slots
            int kk  = (id >> 5) << 1;             // even k
            int n4  = (id & 31) << 2;
            const float* wp = W + (size_t)(k0 + kk) * N + n_blk + n4;
            rb[r][0] = *(const float4*)wp;
            rb[r][1] = *(const float4*)(wp + N);
        }
    };

    // ---- registers -> LDS (fp32 -> packed bf16x2) ----
    auto store_tiles = [&](int buf) {
        unsigned int* a = sA[buf];
        unsigned int* b = sB[buf];
#pragma unroll
        for (int r = 0; r < 4; ++r) {
            int id  = tid + 256 * r;
            int row = id >> 3;
            int k2  = (id & 7) << 1;              // dword index along k (even)
            float4 v = ra[r];
            if (ADD2) {
                v.x += ra2[r].x; v.y += ra2[r].y; v.z += ra2[r].z; v.w += ra2[r].w;
            }
            // 8B-aligned pair -> ds_store_b64
            *(uint2*)(a + row * LDSW + k2) =
                make_uint2(pk_bf16(v.x, v.y), pk_bf16(v.z, v.w));
        }
#pragma unroll
        for (int r = 0; r < 2; ++r) {
            int id  = tid + 256 * r;
            int kk2 = id >> 5;                    // dword index along k
            int n4  = (id & 31) << 2;
            float4 lo = rb[r][0], hi = rb[r][1];  // k and k+1 rows of W
            b[(n4 + 0) * LDSW + kk2] = pk_bf16(lo.x, hi.x);
            b[(n4 + 1) * LDSW + kk2] = pk_bf16(lo.y, hi.y);
            b[(n4 + 2) * LDSW + kk2] = pk_bf16(lo.z, hi.z);
            b[(n4 + 3) * LDSW + kk2] = pk_bf16(lo.w, hi.w);
        }
    };

    // ---- fragment loads (ISA 7.12.2 16-bit layouts, ds_load_b128) + 8x WMMA ----
    auto compute = [&](int buf) {
        const unsigned int* a = sA[buf];
        const unsigned int* b = sB[buf];
        Frag16 af[2], bfr[4];
#pragma unroll
        for (int i = 0; i < 2; ++i) {
            int row = wm * 32 + i * 16 + lr;
            const uint4* p = (const uint4*)(a + row * LDSW + 4 * kh);
            af[i].q[0] = p[0];    // K = 0..7   (+8*kh), dwords 0-3
            af[i].q[1] = p[2];    // K = 16..23 (+8*kh), dwords 8-11
        }
#pragma unroll
        for (int j = 0; j < 4; ++j) {
            int col = wn * 64 + j * 16 + lr;
            const uint4* p = (const uint4*)(b + col * LDSW + 4 * kh);
            bfr[j].q[0] = p[0];   // K = 0..7   (+8*kh)
            bfr[j].q[1] = p[2];   // K = 16..23 (+8*kh)
        }
#pragma unroll
        for (int i = 0; i < 2; ++i)
#pragma unroll
            for (int j = 0; j < 4; ++j)
                acc[i][j] = __builtin_amdgcn_wmma_f32_16x16x32_bf16(
                    false, af[i].bf, false, bfr[j].bf,
                    (short)0, acc[i][j], false, false);
    };

    // ---- double-buffered main loop: one barrier per K-step ----
    const int nsteps = K >> 5;
    load_tiles(0);
    store_tiles(0);
    __syncthreads();
    for (int t = 0; t < nsteps; ++t) {
        const int cur = t & 1;
        if (t + 1 < nsteps) load_tiles((t + 1) << 5);   // global loads overlap WMMA
        compute(cur);
        if (t + 1 < nsteps) {
            store_tiles(cur ^ 1);
            __syncthreads();
        }
    }

    // ---- epilogue: bias / accumulate / relu / store ----
#pragma unroll
    for (int i = 0; i < 2; ++i) {
#pragma unroll
        for (int j = 0; j < 4; ++j) {
            int col = n_blk + wn * 64 + j * 16 + lr;
#pragma unroll
            for (int v = 0; v < 8; ++v) {
                int row = m_blk + wm * 32 + i * 16 + kh * 8 + v;
                if (row < M) {
                    float* cp = C + (size_t)row * N + col;
                    float val = acc[i][j][v];
                    if (ACCUM)      val += *cp;
                    else if (bias)  val += bias[col];
                    if (RELU)       val = val > 0.f ? val : 0.f;
                    *cp = val;
                }
            }
        }
    }
}

// ---------------------------------------------------------------------------
// agg[dst] += h[src] over edges: one block (128 threads) per edge, float4 per
// thread, hardware f32 atomics (global_atomic_add_f32, no return). h and agg
// both fit in the 192MB L2, so this runs at L2 speed.
// ---------------------------------------------------------------------------
__global__ __launch_bounds__(128)
void scatter_add_edges(const float* __restrict__ h, const long long* __restrict__ ei,
                       float* __restrict__ agg, int n_edges)
{
    int e = blockIdx.x;
    if (e >= n_edges) return;
    long long s = ei[e];
    long long d = ei[(size_t)n_edges + e];
    float4 v = *(const float4*)(h + (size_t)s * HID + threadIdx.x * 4);
    float* ap = agg + (size_t)d * HID + threadIdx.x * 4;
    unsafeAtomicAdd(ap + 0, v.x);
    unsafeAtomicAdd(ap + 1, v.y);
    unsafeAtomicAdd(ap + 2, v.z);
    unsafeAtomicAdd(ap + 3, v.w);
}

__global__ void zero_f4(float4* p, int n4)
{
    int i = blockIdx.x * blockDim.x + threadIdx.x;
    if (i < n4) p[i] = make_float4(0.f, 0.f, 0.f, 0.f);
}

extern "C" void kernel_launch(void* const* d_in, const int* in_sizes, int n_in,
                              void* d_out, int out_size, void* d_ws, size_t ws_size,
                              hipStream_t stream)
{
    const float*     x       = (const float*)d_in[0];
    const long long* ei      = (const long long*)d_in[1];
    const float*     in_W    = (const float*)d_in[2];
    const float*     in_b    = (const float*)d_in[3];
    const float*     conv_W1 = (const float*)d_in[4];
    const float*     conv_b1 = (const float*)d_in[5];
    const float*     conv_W2 = (const float*)d_in[6];
    const float*     conv_b2 = (const float*)d_in[7];
    const float*     out_W   = (const float*)d_in[8];
    const float*     out_b   = (const float*)d_in[9];
    float*           out     = (float*)d_out;

    const int M       = NNODES;
    const int n_edges = in_sizes[1] / 2;

    const size_t nh = (size_t)M * HID;          // 25.6M floats per buffer
    float* bufH = (float*)d_ws;                 // current hidden
    float* bufT = bufH + nh;                    // agg / next hidden (ping-pong)
    float* bufZ = bufT + nh;                    // MLP intermediate

    dim3 blk(256);
    dim3 gH(HID / 128,     (M + 127) / 128);    // N=512 GEMMs
    dim3 gO(OUT_DIM / 128, (M + 127) / 128);    // N=256 GEMMs

    // h0 = x @ in_W + in_b
    gemm_bf16_wmma<false, false, false><<<gH, blk, 0, stream>>>(
        x, nullptr, in_W, in_b, bufH, M, HID, IN_DIM);
    // out = h0 @ out_W[0:512,:] + out_b   (initializes d_out)
    gemm_bf16_wmma<false, false, false><<<gO, blk, 0, stream>>>(
        bufH, nullptr, out_W, out_b, out, M, OUT_DIM, HID);

    float* hcur  = bufH;
    float* hnext = bufT;
    const int n4 = (int)(nh / 4);
    for (int l = 0; l < NLAYERS; ++l) {
        // agg = segment_sum(h[src], dst)
        zero_f4<<<(n4 + 255) / 256, 256, 0, stream>>>((float4*)hnext, n4);
        scatter_add_edges<<<n_edges, 128, 0, stream>>>(hcur, ei, hnext, n_edges);
        // z = relu((h + agg) @ W1[l] + b1[l])
        gemm_bf16_wmma<true, true, false><<<gH, blk, 0, stream>>>(
            hcur, hnext, conv_W1 + (size_t)l * HID * HID, conv_b1 + l * HID,
            bufZ, M, HID, HID);
        // h_next = z @ W2[l] + b2[l]   (overwrites agg buffer)
        gemm_bf16_wmma<false, false, false><<<gH, blk, 0, stream>>>(
            bufZ, nullptr, conv_W2 + (size_t)l * HID * HID, conv_b2 + l * HID,
            hnext, M, HID, HID);
        // out += h_next @ out_W[(l+1)*512 : (l+2)*512, :]
        gemm_bf16_wmma<false, false, true><<<gO, blk, 0, stream>>>(
            hnext, nullptr, out_W + (size_t)(l + 1) * HID * OUT_DIM, nullptr,
            out, M, OUT_DIM, HID);
        float* t = hcur; hcur = hnext; hnext = t;
    }
}
